// Head_3745211482460
// MI455X (gfx1250) — compile-verified, matrix operand order checked by reading, not run on hardware
//
#include <hip/hip_runtime.h>

// CDNA5 wave32 WMMA flash attention (bf16 compute, f32 accumulate).
typedef __attribute__((ext_vector_type(16))) __bf16 v16bf;
typedef __attribute__((ext_vector_type(8)))  float  v8f;
typedef __attribute__((ext_vector_type(4)))  __bf16 v4bf;
typedef __attribute__((ext_vector_type(4)))  int    v4i;

#define NB 4
#define NT 4096
#define NC 1024
#define NH 64
#define SCALE 0.03125f   // 1024^-0.5

#define AS1 __attribute__((address_space(1)))
#define AS3 __attribute__((address_space(3)))

// Async global->LDS copy (GLOBAL_LOAD_ASYNC_TO_LDS_B128, ASYNCcnt-tracked).
// Builtin signature (from probe diagnostic): (AS1 v4i*, AS3 v4i*, imm off, imm cpol).
__device__ __forceinline__ void async_copy_b128(const void* g, void* l) {
#if __has_builtin(__builtin_amdgcn_global_load_async_to_lds_b128)
  __builtin_amdgcn_global_load_async_to_lds_b128(
      (AS1 v4i*)(g), (AS3 v4i*)(l), 0, 0);
#else
  *(float4*)l = *(const float4*)g;
#endif
}

__device__ __forceinline__ void wait_async() {
#if __has_builtin(__builtin_amdgcn_global_load_async_to_lds_b128)
#if __has_builtin(__builtin_amdgcn_s_wait_asynccnt)
  __builtin_amdgcn_s_wait_asynccnt(0);
#else
  asm volatile("s_wait_asynccnt 0" ::: "memory");
#endif
#endif
}

// CDNA5 ISA 7.12.2: 16-bit A/B fragment. Elements j=0..7 -> K = laneHi*8 + j,
// elements j=8..15 -> K = 16 + laneHi*8 + (j-8). Both runs are contiguous, so
// a fragment whose K axis is contiguous in LDS is exactly two b128 loads.
union Frag16 { v16bf v; float4 f4[2]; };

__device__ __forceinline__ v16bf load_frag_row(const __bf16* rowPtr, int laneHi) {
  Frag16 u;
  u.f4[0] = *(const float4*)(rowPtr + laneHi * 8);        // K = laneHi*8 .. +7
  u.f4[1] = *(const float4*)(rowPtr + 16 + laneHi * 8);   // K = 16+laneHi*8 .. +7
  return u.v;
}

// ---------------- Kernel 0: one-time weight transpose fp32 -> bf16 ----------
// wT layout: [3*64][1024]  (row = w*64 + out_col, col = k). 384 KB, L2-resident.
__global__ __launch_bounds__(256) void wtrans_kernel(
    const float* __restrict__ Wk, const float* __restrict__ Wq,
    const float* __restrict__ Wv, __bf16* __restrict__ wT)
{
  int idx = blockIdx.x * 256 + threadIdx.x;  // 0 .. 3*1024*64-1
  int c = idx & 63;
  int k = (idx >> 6) & 1023;
  int w = idx >> 16;
  const float* W = (w == 0) ? Wk : (w == 1) ? Wq : Wv;
  wT[(size_t)(w * 64 + c) * NC + k] = (__bf16)W[(size_t)k * NH + c];
}

// ---------------- Kernel 1: QKV projection  y = x @ W -----------------------
// grid = 256 blocks (64 rows each), block = 128 threads = 4 waves.
// Outputs: k,q row-major [16384][64] bf16;  v transposed [B][64][4096] bf16.
__global__ __launch_bounds__(128) void qkv_proj_kernel(
    const float* __restrict__ x, const __bf16* __restrict__ wT,
    __bf16* __restrict__ kq, __bf16* __restrict__ qq, __bf16* __restrict__ vT)
{
  __shared__ __align__(16) __bf16 xa[64][40];    // 64 rows x 32 K (80B stride)
  __shared__ __align__(16) __bf16 wl[192][40];   // 192 out-cols x 32 K

  const int tid  = threadIdx.x;
  const int lane = tid & 31;
  const int laneLo = lane & 15, laneHi = (lane >> 4) & 1;
  const int wave = tid >> 5;
  const int rowBase = blockIdx.x * 64;

  v8f acc[12];
#pragma unroll
  for (int i = 0; i < 12; ++i)
#pragma unroll
    for (int r = 0; r < 8; ++r) acc[i][r] = 0.0f;

  for (int kk = 0; kk < NC; kk += 32) {
    // Stage x tile 64x32 fp32 -> bf16 (converted in flight, packed b64 stores).
#pragma unroll
    for (int i = 0; i < 4; ++i) {
      int idx = tid + i * 128;              // 512 float4 chunks
      int r = idx >> 3, c = (idx & 7) << 2;
      float4 f = *(const float4*)(x + (size_t)(rowBase + r) * NC + kk + c);
      v4bf t = { (__bf16)f.x, (__bf16)f.y, (__bf16)f.z, (__bf16)f.w };
      *(v4bf*)&xa[r][c] = t;
    }
    // Stage wT slice (pure copy): async global->LDS, ASYNCcnt-tracked.
#pragma unroll
    for (int i = 0; i < 6; ++i) {
      int idx = tid + i * 128;              // 768 16B chunks
      int row = idx >> 2, c = (idx & 3) << 3;
      async_copy_b128(wT + (size_t)row * NC + kk + c, &wl[row][c]);
    }
    wait_async();
    __syncthreads();

    v16bf a = load_frag_row(&xa[wave * 16 + laneLo][0], laneHi);
#pragma unroll
    for (int nt = 0; nt < 12; ++nt) {
      v16bf bm = load_frag_row(&wl[nt * 16 + laneLo][0], laneHi);
      acc[nt] = __builtin_amdgcn_wmma_f32_16x16x32_bf16(
          false, a, false, bm, (short)0, acc[nt], false, false);
    }
    __syncthreads();
  }

  const int laneHi8 = laneHi << 3;
  // k, q: row-major. C/D layout: elem r -> row r+8*laneHi, col laneLo+16*t.
#pragma unroll
  for (int t = 0; t < 4; ++t)
#pragma unroll
    for (int r = 0; r < 8; ++r) {
      int row = rowBase + wave * 16 + r + laneHi8;
      kq[(size_t)row * NH + t * 16 + laneLo] = (__bf16)acc[t][r];
      qq[(size_t)row * NH + t * 16 + laneLo] = (__bf16)acc[4 + t][r];
    }
  // v: transposed [B][64][4096] so the PV B-fragment is row-contiguous later.
#pragma unroll
  for (int t = 0; t < 4; ++t)
#pragma unroll
    for (int r = 0; r < 8; ++r) {
      int row = rowBase + wave * 16 + r + laneHi8;
      int bb = row >> 12, tt = row & 4095;
      vT[((size_t)bb * NH + t * 16 + laneLo) * NT + tt] = (__bf16)acc[8 + t][r];
    }
}

// ---------------- Kernel 2: flash attention ---------------------------------
// grid = B * (T/64) = 256 blocks, block = 128 threads = 4 waves,
// each wave owns 16 query rows; 64-key tiles staged via async LDS copies.
__global__ __launch_bounds__(128) void attn_kernel(
    const __bf16* __restrict__ qg, const __bf16* __restrict__ kg_,
    const __bf16* __restrict__ vT, float* __restrict__ out)
{
  __shared__ __align__(16) __bf16 kt[64][72];     // K tile rows (144B stride)
  __shared__ __align__(16) __bf16 vtT[64][72];    // V^T tile: [hd][key]
  __shared__ __align__(16) __bf16 pt[4][16][72];  // per-wave P (C->A relayout)

  const int tid  = threadIdx.x;
  const int lane = tid & 31;
  const int laneLo = lane & 15, laneHi = (lane >> 4) & 1, laneHi8 = laneHi << 3;
  const int wave = tid >> 5;
  const int b    = blockIdx.x >> 6;
  const int qt   = blockIdx.x & 63;
  const int qBase = qt * 64;
  const size_t base = (size_t)b * NT * NH;

  // Stage Q tile into kt, build resident A fragments (head dim = 2 K-steps).
#pragma unroll
  for (int i = 0; i < 4; ++i) {
    int idx = tid + i * 128;
    int r = idx >> 3, c = (idx & 7) << 3;
    async_copy_b128(qg + base + (size_t)(qBase + r) * NH + c, &kt[r][c]);
  }
  wait_async();
  __syncthreads();
  v16bf qf[2];
#pragma unroll
  for (int st = 0; st < 2; ++st)
    qf[st] = load_frag_row(&kt[wave * 16 + laneLo][st * 32], laneHi);
  __syncthreads();

  v8f o[4];
  float m[8], l[8];
#pragma unroll
  for (int nt = 0; nt < 4; ++nt)
#pragma unroll
    for (int r = 0; r < 8; ++r) o[nt][r] = 0.0f;
#pragma unroll
  for (int r = 0; r < 8; ++r) { m[r] = -1e30f; l[r] = 0.0f; }

  for (int jt = 0; jt <= qt; ++jt) {
    // Stage K tile (row-major keys) and V^T tile ([hd][key]) asynchronously.
#pragma unroll
    for (int i = 0; i < 4; ++i) {
      int idx = tid + i * 128;
      int r = idx >> 3, c = (idx & 7) << 3;
      async_copy_b128(kg_ + base + (size_t)(jt * 64 + r) * NH + c, &kt[r][c]);
      async_copy_b128(vT + ((size_t)b * NH + r) * NT + jt * 64 + c, &vtT[r][c]);
    }
    wait_async();
    __syncthreads();

    // S = Q K^T  (4 key sub-tiles x 2 head-dim steps)
    v8f s[4];
#pragma unroll
    for (int nsub = 0; nsub < 4; ++nsub) {
#pragma unroll
      for (int r = 0; r < 8; ++r) s[nsub][r] = 0.0f;
#pragma unroll
      for (int st = 0; st < 2; ++st) {
        v16bf bm = load_frag_row(&kt[nsub * 16 + laneLo][st * 32], laneHi);
        s[nsub] = __builtin_amdgcn_wmma_f32_16x16x32_bf16(
            false, qf[st], false, bm, (short)0, s[nsub], false, false);
      }
    }

    // Scale + causal mask (elem: row = r+8*laneHi, key = 16*nsub+laneLo).
    float sm[4][8];
#pragma unroll
    for (int nsub = 0; nsub < 4; ++nsub)
#pragma unroll
      for (int r = 0; r < 8; ++r) {
        int key = jt * 64 + nsub * 16 + laneLo;
        int row = qBase + wave * 16 + r + laneHi8;
        sm[nsub][r] = (key <= row) ? s[nsub][r] * SCALE : -1e30f;
      }

    // Online softmax: row stats live across the 16-lane half (C-layout).
#pragma unroll
    for (int r = 0; r < 8; ++r) {
      float t = fmaxf(fmaxf(sm[0][r], sm[1][r]), fmaxf(sm[2][r], sm[3][r]));
      t = fmaxf(t, __shfl_xor(t, 1, 32));
      t = fmaxf(t, __shfl_xor(t, 2, 32));
      t = fmaxf(t, __shfl_xor(t, 4, 32));
      t = fmaxf(t, __shfl_xor(t, 8, 32));
      float mn = fmaxf(m[r], t);
      float alpha = __expf(m[r] - mn);
      m[r] = mn;
      float rs = 0.0f;
#pragma unroll
      for (int nsub = 0; nsub < 4; ++nsub) {
        float p = __expf(sm[nsub][r] - mn);
        sm[nsub][r] = p;
        rs += p;
      }
      rs += __shfl_xor(rs, 1, 32);
      rs += __shfl_xor(rs, 2, 32);
      rs += __shfl_xor(rs, 4, 32);
      rs += __shfl_xor(rs, 8, 32);
      l[r] = l[r] * alpha + rs;
#pragma unroll
      for (int nt = 0; nt < 4; ++nt) o[nt][r] *= alpha;
    }

    // Relayout P (C-layout f32 -> A-layout bf16) through per-wave LDS.
#pragma unroll
    for (int nsub = 0; nsub < 4; ++nsub)
#pragma unroll
      for (int r = 0; r < 8; ++r)
        pt[wave][r + laneHi8][nsub * 16 + laneLo] = (__bf16)sm[nsub][r];
    asm volatile("s_wait_dscnt 0" ::: "memory");  // wave-local store->load order

    // O += P V   (2 key steps x 4 head-dim sub-tiles)
#pragma unroll
    for (int st = 0; st < 2; ++st) {
      v16bf pa = load_frag_row(&pt[wave][laneLo][st * 32], laneHi);
#pragma unroll
      for (int nt = 0; nt < 4; ++nt) {
        v16bf vb = load_frag_row(&vtT[nt * 16 + laneLo][st * 32], laneHi);
        o[nt] = __builtin_amdgcn_wmma_f32_16x16x32_bf16(
            false, pa, false, vb, (short)0, o[nt], false, false);
      }
    }
    __syncthreads();  // protect kt/vtT before next tile overwrites them
  }

  // Epilogue: O * (1/l) -> fp32 output [B,T,H]
#pragma unroll
  for (int r = 0; r < 8; ++r) {
    float rl = 1.0f / l[r];
    int row = qBase + wave * 16 + r + laneHi8;
#pragma unroll
    for (int nt = 0; nt < 4; ++nt)
      out[base + (size_t)row * NH + nt * 16 + laneLo] = o[nt][r] * rl;
  }
}

extern "C" void kernel_launch(void* const* d_in, const int* in_sizes, int n_in,
                              void* d_out, int out_size, void* d_ws, size_t ws_size,
                              hipStream_t stream) {
  const float* x  = (const float*)d_in[0];
  const float* Wk = (const float*)d_in[1];
  const float* Wq = (const float*)d_in[2];
  const float* Wv = (const float*)d_in[3];
  float* out = (float*)d_out;

  const size_t elems = (size_t)NB * NT * NH;  // 1M elements = 2MB bf16
  __bf16* q_ws = (__bf16*)d_ws;
  __bf16* k_ws = q_ws + elems;
  __bf16* vT   = k_ws + elems;                // [B][64][4096]
  __bf16* wT   = vT + elems;                  // [192][1024]

  wtrans_kernel<<<(3 * NC * NH) / 256, 256, 0, stream>>>(Wk, Wq, Wv, wT);
  qkv_proj_kernel<<<(NB * NT) / 64, 128, 0, stream>>>(x, wT, k_ws, q_ws, vT);
  attn_kernel<<<NB * (NT / 64), 128, 0, stream>>>(q_ws, k_ws, vT, out);
}
